// GATGraphEncoder_31782757991166
// MI455X (gfx1250) — matrix operand order repaired and checked by reading, not compiled, not run on hardware
//
#include <hip/hip_runtime.h>
#include <hip/hip_bf16.h>

// ---------------------------------------------------------------------------
// Sizes (fixed by the reference)
// ---------------------------------------------------------------------------
#define NN   30000
#define EE   480000
#define FF   128      // 2*IN == HEADS*HID == head din == head dout
#define TPB  256

typedef __attribute__((ext_vector_type(16))) __bf16 v16bf;
typedef __attribute__((ext_vector_type(8)))  float  v8f;

// ---------------------------------------------------------------------------
// Helpers
// ---------------------------------------------------------------------------
__device__ __forceinline__ float leaky(float x) { return x > 0.0f ? x : 0.2f * x; }

__device__ __forceinline__ unsigned f2bf_bits(float f) {
    unsigned u = __float_as_uint(f);
    unsigned r = u + 0x7FFFu + ((u >> 16) & 1u);   // round-to-nearest-even
    return r >> 16;
}

// monotone float <-> int encoding so atomicMax(int) == float max
__device__ __forceinline__ int encf(float f) {
    int i = __float_as_int(f);
    return i >= 0 ? i : (i ^ 0x7FFFFFFF);
}
__device__ __forceinline__ float decf(int i) {
    return __int_as_float(i >= 0 ? i : (i ^ 0x7FFFFFFF));
}

// ---------------------------------------------------------------------------
// 1. xf = concat(x, emb[node_type])  (f32 + bf16 copies)
// ---------------------------------------------------------------------------
__global__ void k_build_xf(const float* __restrict__ x, const int* __restrict__ nt,
                           const float* __restrict__ emb,
                           float* __restrict__ xf, unsigned short* __restrict__ xf16) {
    int i = blockIdx.x * blockDim.x + threadIdx.x;
    if (i >= NN * FF) return;
    int n = i >> 7, c = i & 127;
    float v = (c < 64) ? x[n * 64 + c] : emb[nt[n] * 64 + (c - 64)];
    xf[i] = v;
    xf16[i] = (unsigned short)f2bf_bits(v);
}

__global__ void k_zero_f32(float* __restrict__ p, int n) {
    int i = blockIdx.x * blockDim.x + threadIdx.x;
    if (i < n) p[i] = 0.0f;
}

// ---------------------------------------------------------------------------
// 2a. Pack W (f32, [128 x 128] row-major, h = xf @ W) into the per-lane WMMA
//     B-fragment image:  dword index = ((ntile*32 + lane)*4 + kb)*8 + p
//     so each lane's whole B operand (all 4 K-steps) is 32 contiguous dwords.
//     B k order chosen here (contiguous, same as A): lane = column (lane&15); lanes 0-15 K=0..15,
//     lanes 16-31 K=16..31; pair p packs K = 2p (lo) and 2p+1 (hi).
// ---------------------------------------------------------------------------
__global__ void k_pack_w(const float* __restrict__ W, unsigned* __restrict__ out) {
    int idx = blockIdx.x * blockDim.x + threadIdx.x;
    if (idx >= 8 * 32 * 4 * 8) return;          // 8192 dwords = 128x128 bf16
    int ntile = idx >> 10;
    int lane  = (idx >> 5) & 31;
    int kb    = (idx >> 3) & 3;
    int p     = idx & 7;
    int col   = ntile * 16 + (lane & 15);
    int kB    = kb * 32 + ((lane >> 4) ? 16 : 0) + 2 * p;
    unsigned lo = f2bf_bits(W[kB * FF + col]);
    unsigned hi = f2bf_bits(W[(kB + 1) * FF + col]);
    out[idx] = lo | (hi << 16);
}

// ---------------------------------------------------------------------------
// 2b. GEMM: C[nrows x 128] = A[nrows x 128](bf16) * B[128 x 128](bf16), f32 acc
//     One wave per 16x16 tile; B fragment = 32 contiguous dwords per lane
//     (pre-packed, L2-broadcast); A per K-step = 2 x b128; 4 x v_wmma.
// ---------------------------------------------------------------------------
__global__ void k_gemm_wmma(const unsigned short* __restrict__ A,
                            const unsigned* __restrict__ Bpk,
                            float* __restrict__ C, int nrows) {
    int wid  = (blockIdx.x * blockDim.x + threadIdx.x) >> 5;
    int lane = threadIdx.x & 31;
    int tiles_m = (nrows + 15) >> 4;
    int mt = wid >> 3;            // 8 N-tiles (128/16)
    int ntile = wid & 7;
    if (mt >= tiles_m) return;
    int half = lane >> 4, l16 = lane & 15;
    int row = mt * 16 + l16;
    int col = ntile * 16 + l16;
    const unsigned* arow = (const unsigned*)(A + (size_t)row * FF);   // 2 bf16/dword
    const unsigned* bln  = Bpk + ((ntile * 32 + lane) << 5);          // 32 dwords

    // load the whole B operand up front (8 x b128, shared across all M-tiles)
    unsigned bu[32];
#pragma unroll
    for (int t = 0; t < 32; ++t) bu[t] = bln[t];

    v8f c = {};
    union Frag { v16bf v; unsigned u[8]; };
#pragma unroll
    for (int kb = 0; kb < 4; ++kb) {
        Frag a, b;
        int kbase = kb * 32;
#pragma unroll
        for (int p = 0; p < 8; ++p) {
            // A k order chosen here (contiguous): lanes 0-15 K=0..15, lanes 16-31 K=16..31 (matches packed B)
            int kA = kbase + 2 * p + (half ? 16 : 0);
            a.u[p] = arow[kA >> 1];
            b.u[p] = bu[kb * 8 + p];
        }
        c = __builtin_amdgcn_wmma_f32_16x16x32_bf16(false, a.v, false, b.v,
                                                    (short)0, c, false, false);
    }
#pragma unroll
    for (int r = 0; r < 8; ++r) {
        int m = mt * 16 + r + (half ? 8 : 0);
        C[(size_t)m * FF + col] = c[r];
    }
}

// ---------------------------------------------------------------------------
// 3. a_s[n,h] = sum_d h[n,h,d]*att_src[h,d]; same for a_d
// ---------------------------------------------------------------------------
__global__ void k_as_ad(const float* __restrict__ h, const float* __restrict__ att_s,
                        const float* __restrict__ att_d,
                        float* __restrict__ a_s, float* __restrict__ a_d,
                        int heads, int dout) {
    int i = blockIdx.x * blockDim.x + threadIdx.x;
    if (i >= NN * heads) return;
    int n = i / heads, hh = i - n * heads;
    const float* hp = h + (size_t)n * FF + hh * dout;
    const float* sp = att_s + hh * dout;
    const float* dp = att_d + hh * dout;
    float s = 0.0f, d = 0.0f;
    for (int j = 0; j < dout; ++j) { s += hp[j] * sp[j]; d += hp[j] * dp[j]; }
    a_s[i] = s; a_d[i] = d;
}

// init m with the (always unmasked) self-loop logit, denom = 0
__global__ void k_init_m(const float* __restrict__ a_s, const float* __restrict__ a_d,
                         int* __restrict__ menc, float* __restrict__ denom, int heads) {
    int i = blockIdx.x * blockDim.x + threadIdx.x;
    if (i >= NN * heads) return;
    menc[i]  = encf(leaky(a_s[i] + a_d[i]));
    denom[i] = 0.0f;
}

// ---------------------------------------------------------------------------
// 4. edge passes (rel < 0 => no relation filtering, used by output head)
// ---------------------------------------------------------------------------
__global__ void k_edge_max(const int* __restrict__ src, const int* __restrict__ dst,
                           const int* __restrict__ et, int rel,
                           const float* __restrict__ a_s, const float* __restrict__ a_d,
                           int* __restrict__ menc, int heads) {
    int e = blockIdx.x * blockDim.x + threadIdx.x;
    if (e >= EE) return;
    if (rel >= 0 && et[e] != rel) return;
    int s = src[e], d = dst[e];
    for (int h = 0; h < heads; ++h) {
        float lg = leaky(a_s[s * heads + h] + a_d[d * heads + h]);
        atomicMax(&menc[d * heads + h], encf(lg));
    }
}

// one wave32 per edge: lanes cover 128 feature columns (4 per lane)
__global__ void k_edge_sum(const int* __restrict__ src, const int* __restrict__ dst,
                           const int* __restrict__ et, int rel,
                           const float* __restrict__ a_s, const float* __restrict__ a_d,
                           const int* __restrict__ menc, float* __restrict__ denom,
                           const float* __restrict__ h, float* __restrict__ num,
                           int heads, int dshift) {
    int gw   = (blockIdx.x * blockDim.x + threadIdx.x) >> 5;
    int lane = threadIdx.x & 31;
    if (gw >= EE) return;
    if (rel >= 0 && et[gw] != rel) return;
    int s = src[gw], d = dst[gw];
    float eh[2];
    for (int hh = 0; hh < heads; ++hh) {
        float lg = leaky(a_s[s * heads + hh] + a_d[d * heads + hh]);
        eh[hh] = __expf(lg - decf(menc[d * heads + hh]));
    }
    if (lane < heads) atomicAdd(&denom[d * heads + lane], eh[lane]);
    const float* hs = h + (size_t)s * FF;
    float* nd = num + (size_t)d * FF;
    for (int c = lane; c < FF; c += 32)
        atomicAdd(&nd[c], eh[c >> dshift] * hs[c]);
}

// ---------------------------------------------------------------------------
// 5. per-node finalize: fold in self loop, normalize, add bias, accumulate
// ---------------------------------------------------------------------------
__global__ void k_finalize_acc(const float* __restrict__ a_s, const float* __restrict__ a_d,
                               const int* __restrict__ menc, const float* __restrict__ denom,
                               const float* __restrict__ h, const float* __restrict__ num,
                               const float* __restrict__ bias, float* __restrict__ acc,
                               int heads, int dshift) {
    int i = blockIdx.x * blockDim.x + threadIdx.x;
    if (i >= NN * FF) return;
    int n = i >> 7, c = i & 127;
    int hh = c >> dshift;
    float md = decf(menc[n * heads + hh]);
    float es = __expf(leaky(a_s[n * heads + hh] + a_d[n * heads + hh]) - md);
    float dt = denom[n * heads + hh] + es;
    acc[i] += (num[i] + es * h[i]) / dt + bias[c];
}

__global__ void k_relu_commit(const float* __restrict__ acc, float* __restrict__ xf,
                              unsigned short* __restrict__ xf16) {
    int i = blockIdx.x * blockDim.x + threadIdx.x;
    if (i >= NN * FF) return;
    float v = acc[i] > 0.0f ? acc[i] : 0.0f;
    xf[i] = v;
    xf16[i] = (unsigned short)f2bf_bits(v);
}

// output head: pooled[c] += ((num + es*h)/dt + bias[c]) / N
__global__ void k_head_finalize(const float* __restrict__ a_s, const float* __restrict__ a_d,
                                const int* __restrict__ menc, const float* __restrict__ denom,
                                const float* __restrict__ h, const float* __restrict__ num,
                                const float* __restrict__ bias, float* __restrict__ pooled) {
    int i = blockIdx.x * blockDim.x + threadIdx.x;
    if (i >= NN * FF) return;
    int n = i >> 7, c = i & 127;
    float md = decf(menc[n]);
    float es = __expf(leaky(a_s[n] + a_d[n]) - md);
    float dt = denom[n] + es;
    float v = (num[i] + es * h[i]) / dt + bias[c];
    atomicAdd(&pooled[c], v * (1.0f / (float)NN));
}

// alpha for every edge + self loop (output head, 1 head, no mask)
__global__ void k_alpha(const int* __restrict__ src, const int* __restrict__ dst,
                        const float* __restrict__ a_s, const float* __restrict__ a_d,
                        const int* __restrict__ menc, const float* __restrict__ denom,
                        float* __restrict__ alpha) {
    int i = blockIdx.x * blockDim.x + threadIdx.x;
    if (i >= EE + NN) return;
    int s, d;
    if (i < EE) { s = src[i]; d = dst[i]; } else { s = d = i - EE; }
    float md = decf(menc[d]);
    float dt = denom[d] + __expf(leaky(a_s[d] + a_d[d]) - md);
    alpha[i] = __expf(leaky(a_s[s] + a_d[d]) - md) / dt;
}

// ---------------------------------------------------------------------------
// Host driver
// ---------------------------------------------------------------------------
extern "C" void kernel_launch(void* const* d_in, const int* in_sizes, int n_in,
                              void* d_out, int out_size, void* d_ws, size_t ws_size,
                              hipStream_t stream) {
    const float* x        = (const float*)d_in[0];
    const int*   node_ty  = (const int*)  d_in[1];
    const int*   ei       = (const int*)  d_in[2];   // [2,E]: src then dst
    const int*   et       = (const int*)  d_in[3];
    const float* emb      = (const float*)d_in[4];
    const int* srcI = ei;
    const int* dstI = ei + EE;

    // params pytree (sorted keys): emb, layers[2][4]{W,att_dst,att_src,bias}, out{...}
    const float *W[9], *Adst[9], *Asrc[9], *Bias[9];
    for (int i = 0; i < 9; ++i) {
        int base = 5 + i * 4;
        W[i]    = (const float*)d_in[base + 0];
        Adst[i] = (const float*)d_in[base + 1];
        Asrc[i] = (const float*)d_in[base + 2];
        Bias[i] = (const float*)d_in[base + 3];
    }

    // workspace carve
    char* wp = (char*)d_ws;
    auto carve = [&](size_t bytes) -> void* {
        void* r = (void*)wp;
        wp += (bytes + 255) & ~(size_t)255;
        return r;
    };
    float*          xf    = (float*)         carve((size_t)NN * FF * 4);
    unsigned short* xf16  = (unsigned short*)carve((size_t)NN * FF * 2);
    float*          hbuf  = (float*)         carve((size_t)NN * FF * 4);
    float*          num   = (float*)         carve((size_t)NN * FF * 4);
    float*          acc   = (float*)         carve((size_t)NN * FF * 4);
    float*          a_s   = (float*)         carve((size_t)NN * 2 * 4);
    float*          a_d   = (float*)         carve((size_t)NN * 2 * 4);
    int*            menc  = (int*)           carve((size_t)NN * 2 * 4);
    float*          denom = (float*)         carve((size_t)NN * 2 * 4);
    unsigned*       wpk   = (unsigned*)      carve((size_t)8192 * 4);  // packed W

    const int NF = NN * FF;
    dim3 bNF((NF + TPB - 1) / TPB), bT(TPB);
    int waves = ((NN + 15) / 16) * 8;
    dim3 bGemm((waves * 32 + TPB - 1) / TPB);
    dim3 bNH2((NN * 2 + TPB - 1) / TPB);
    dim3 bNH1((NN + TPB - 1) / TPB);
    dim3 bE((EE + TPB - 1) / TPB);
    dim3 bEw((EE * 32 + TPB - 1) / TPB);     // one wave per edge
    dim3 bPk((8192 + TPB - 1) / TPB);
    dim3 bAl((EE + NN + TPB - 1) / TPB);

    k_build_xf<<<bNF, bT, 0, stream>>>(x, node_ty, emb, xf, xf16);

    for (int l = 0; l < 2; ++l) {
        k_zero_f32<<<bNF, bT, 0, stream>>>(acc, NF);
        for (int r = 0; r < 4; ++r) {
            int ci = l * 4 + r;
            k_pack_w<<<bPk, bT, 0, stream>>>(W[ci], wpk);
            k_gemm_wmma<<<bGemm, bT, 0, stream>>>(xf16, wpk, hbuf, NN);
            k_as_ad<<<bNH2, bT, 0, stream>>>(hbuf, Asrc[ci], Adst[ci], a_s, a_d, 2, 64);
            k_init_m<<<bNH2, bT, 0, stream>>>(a_s, a_d, menc, denom, 2);
            k_zero_f32<<<bNF, bT, 0, stream>>>(num, NF);
            k_edge_max<<<bE, bT, 0, stream>>>(srcI, dstI, et, r, a_s, a_d, menc, 2);
            k_edge_sum<<<bEw, bT, 0, stream>>>(srcI, dstI, et, r, a_s, a_d, menc, denom,
                                               hbuf, num, 2, 6);
            k_finalize_acc<<<bNF, bT, 0, stream>>>(a_s, a_d, menc, denom, hbuf, num,
                                                   Bias[ci], acc, 2, 6);
        }
        k_relu_commit<<<bNF, bT, 0, stream>>>(acc, xf, xf16);
    }

    // output head (1 head, dout=128, no relation mask)
    k_pack_w<<<bPk, bT, 0, stream>>>(W[8], wpk);
    k_gemm_wmma<<<bGemm, bT, 0, stream>>>(xf16, wpk, hbuf, NN);
    k_as_ad<<<bNH1, bT, 0, stream>>>(hbuf, Asrc[8], Adst[8], a_s, a_d, 1, 128);
    k_init_m<<<bNH1, bT, 0, stream>>>(a_s, a_d, menc, denom, 1);
    k_zero_f32<<<bNF, bT, 0, stream>>>(num, NF);
    k_edge_max<<<bE, bT, 0, stream>>>(srcI, dstI, et, -1, a_s, a_d, menc, 1);
    k_edge_sum<<<bEw, bT, 0, stream>>>(srcI, dstI, et, -1, a_s, a_d, menc, denom,
                                       hbuf, num, 1, 7);
    float* pooled = (float*)d_out;
    k_zero_f32<<<dim3(1), bT, 0, stream>>>(pooled, FF);
    k_head_finalize<<<bNF, bT, 0, stream>>>(a_s, a_d, menc, denom, hbuf, num,
                                            Bias[8], pooled);
    k_alpha<<<bAl, bT, 0, stream>>>(srcI, dstI, a_s, a_d, menc, denom, pooled + FF);
}
